// MapDetectionLoss_45226005627160
// MI455X (gfx1250) — matrix operand order repaired and checked by reading, not compiled, not run on hardware
//
#include <hip/hip_runtime.h>

typedef __attribute__((ext_vector_type(16))) _Float16 v16h;
typedef __attribute__((ext_vector_type(2)))  _Float16 h2;
typedef __attribute__((ext_vector_type(8)))  float    v8f;
typedef __attribute__((ext_vector_type(8)))  unsigned v8u;
typedef __attribute__((ext_vector_type(4)))  unsigned u32x4;
typedef __attribute__((ext_vector_type(8)))  int      i32x8;
typedef __attribute__((ext_vector_type(4)))  int      i32x4;

#define B_    4
#define N_    1024
#define G_    128
#define P_    20
#define C_    3
#define NTILE 128       // n per block (8 waves x 16 rows)
#define GTILE 16        // g per block
#define NCHUNK P_       // K padded to 32 per p-row -> chunk index == p

#if defined(__has_builtin)
#if __has_builtin(__builtin_amdgcn_tensor_load_to_lds)
#define HAVE_TDM 1
#endif
#endif

__launch_bounds__(256)
__global__ void matcher_cost_kernel(const float* __restrict__ pred_logits,
                                    const float* __restrict__ pred_lines,
                                    const int*   __restrict__ gt_labels,
                                    const float* __restrict__ gt_lines,
                                    const int*   __restrict__ gt_masks,
                                    float* __restrict__ out)
{
    __shared__ float    s_pred[NTILE * P_ * 2];   // [nl][p][xy]  5120 f = 20KB
    __shared__ float    s_logit[NTILE * 4];       // [nl][c] padded
    __shared__ float    s_gt[GTILE * P_ * 2];     // [gi][q][xy]
    __shared__ unsigned s_maskbits[GTILE];
    __shared__ float    s_norminv[GTILE];
    __shared__ int      s_label[GTILE];

    const int tid   = threadIdx.x;
    const int b     = blockIdx.z;
    const int nBase = blockIdx.x * NTILE;
    const int gBase = blockIdx.y * GTILE;

    const float* predg = pred_lines  + (size_t)(b * N_ + nBase) * (P_ * 2);
    const float* logg  = pred_logits + (size_t)(b * N_ + nBase) * C_;
    const float* gtg   = gt_lines    + (size_t)(b * G_ + gBase) * (P_ * 2);

    __builtin_prefetch(logg + tid, 0, 1);       // global_prefetch_b8
    __builtin_prefetch(gtg  + tid * 2, 0, 1);

    // ---- stage pred tile via Tensor Data Mover (wave 0 issues one DMA) ----
#ifdef HAVE_TDM
    if ((tid >> 5) == 0) {
        unsigned           lds_off = (unsigned)(size_t)(&s_pred[0]);
        unsigned long long ga      = (unsigned long long)(size_t)predg;
        // D# group0: count=1 | lds_addr | global_addr | type=2 ("image")
        u32x4 g0 = { 1u,
                     lds_off,
                     (unsigned)(ga & 0xffffffffu),
                     ((unsigned)((ga >> 32) & 0x01ffffffu)) | (2u << 30) };
        // D# group1: data_size=4B; tensor_dim0=5120; tensor_dim1=1;
        //            tile_dim0=5120; tile_dim1=1; dim0_stride=5120
        i32x8 g1 = { (int)(2u << 16),
                     (int)(5120u << 16),
                     (int)(1u << 16),
                     (int)(5120u << 16),
                     1,
                     5120,
                     0, 0 };
        i32x4 gz = { 0, 0, 0, 0 };
#if __clang_major__ >= 23
        i32x8 gz8 = { 0, 0, 0, 0, 0, 0, 0, 0 };
        __builtin_amdgcn_tensor_load_to_lds(g0, g1, gz, gz, gz8, 0);
#else
        __builtin_amdgcn_tensor_load_to_lds(g0, g1, gz, gz, 0);
#endif
        __builtin_amdgcn_s_wait_tensorcnt(0);   // s_wait_tensorcnt 0 before barrier
    }
#else
    for (int i = tid; i < NTILE * P_ * 2; i += 256) s_pred[i] = predg[i];
#endif

    // ---- stage the small per-block data ----
    for (int i = tid; i < NTILE * C_; i += 256) {
        int r = i / C_, c = i - r * C_;
        s_logit[r * 4 + c] = logg[i];
    }
    for (int i = tid; i < GTILE * P_ * 2; i += 256) s_gt[i] = gtg[i];
    if (tid < GTILE) {
        const int* mrow = gt_masks + (size_t)(b * G_ + gBase + tid) * P_;
        unsigned bits = 0; int nv = 0;
        for (int p = 0; p < P_; ++p) { int mv = mrow[p] & 1; bits |= (unsigned)mv << p; nv += mv; }
        s_maskbits[tid] = bits;
        s_norminv[tid]  = 1.0f / (float)(nv > 0 ? nv : 1);
        s_label[tid]    = gt_labels[b * G_ + gBase + tid];
    }
    __syncthreads();

    const int lane = tid & 31;
    const int wave = tid >> 5;       // 0..7 -> 16-row n sub-tile
    const int col  = lane & 15;      // A row M / B,C column
    const int half = lane >> 4;
    const int nSub = wave * 16;
    const float* myPred = &s_pred[(nSub + col) * (P_ * 2)];   // A row M = lane%16

    // Per-lane A-element -> q mapping (within chunk p: q = 16h+i).
    // q>=20 slots hold harmless finite garbage: their B rows are always zero.
    int qmap[16];
    #pragma unroll
    for (int i = 0; i < 16; ++i) {
        int q = 16 * half + i;
        qmap[i] = (q < P_) ? q : 0;
    }

    for (int gi = 0; gi < GTILE; ++gi) {
        const unsigned mbits = s_maskbits[gi];
        const unsigned mb2   = (col <= 4) ? mbits : 0u;   // B2 enable folded into bits
        const float*   gt    = &s_gt[gi * P_ * 2];

        // preload this lane's gt coordinates once per g as packed half2 (x,y)
        h2 gq2[16];
        #pragma unroll
        for (int i = 0; i < 16; ++i) {
            h2 t;
            t.x = (_Float16)gt[2 * qmap[i] + 0];
            t.y = (_Float16)gt[2 * qmap[i] + 1];
            gq2[i] = t;
        }

        v8f acc1 = {};   // columns = shifts 0..15
        v8f acc2 = {};   // cols 0..3 = shifts 16..19, col 4 = reversed polyline

        #pragma unroll 5
        for (int ch = 0; ch < NCHUNK; ++ch) {
            // one pred point per chunk (p == ch): single ds_load_b64 + one cvt_pk
            h2 pa;
            pa.x = (_Float16)myPred[2 * ch + 0];
            pa.y = (_Float16)myPred[2 * ch + 1];

            // A fragment: D[ch][q] = |dx|+|dy| in packed f16
            v16h a;
            #pragma unroll
            for (int i = 0; i < 16; ++i) {
                h2 d = pa - gq2[i];                                    // v_pk_add_f16 (neg)
                d = __builtin_bit_cast(h2,
                        __builtin_bit_cast(unsigned, d) & 0x7FFF7FFFu); // packed abs
                a[i] = d.x + d.y;                                      // v_add_f16 opsel
            }

            // B fragments: one nonzero half per column -> 8 one-hot dwords (branch-free)
            int t1 = ch + col;            t1 -= (t1 >= P_) ? P_ : 0;   // shift s=col
            unsigned v32_1 = ((mbits >> t1) & 1u) ? (0x3C00u << ((t1 & 1) << 4)) : 0u;
            int wi1 = (t1 - 16 * half) >> 1;

            int tl = ch + 16 + col;       tl -= (tl >= P_) ? P_ : 0;   // shifts 16..19
            int t2 = (col == 4) ? (P_ - 1 - ch) : tl;                  // col4 = reversed
            unsigned v32_2 = ((mb2 >> t2) & 1u) ? (0x3C00u << ((t2 & 1) << 4)) : 0u;
            int wi2 = (t2 - 16 * half) >> 1;

            v8u bw1, bw2;
            #pragma unroll
            for (int w = 0; w < 8; ++w) {
                bw1[w] = (w == wi1) ? v32_1 : 0u;
                bw2[w] = (w == wi2) ? v32_2 : 0u;
            }
            v16h b1 = __builtin_bit_cast(v16h, bw1);
            v16h b2 = __builtin_bit_cast(v16h, bw2);

            acc1 = __builtin_amdgcn_wmma_f32_16x16x32_f16(false, a, false, b1,
                                                          (short)0, acc1, false, false);
            acc2 = __builtin_amdgcn_wmma_f32_16x16x32_f16(false, a, false, b2,
                                                          (short)0, acc2, false, false);
        }

        // ---- epilogue ----
        const int   label = __builtin_amdgcn_readfirstlane(s_label[gi]);  // uniform
        const float ninv  = s_norminv[gi];

        #pragma unroll
        for (int j = 0; j < 8; ++j) {
            float ptsRaw;
            if (label == 1) {
                // polygon: min over shifts 0..19
                float v1 = acc1[j];
                float v2 = acc2[j];
                v1 = fminf(v1, __shfl_xor(v1, 1, 16));
                v1 = fminf(v1, __shfl_xor(v1, 2, 16));
                v1 = fminf(v1, __shfl_xor(v1, 4, 16));
                v1 = fminf(v1, __shfl_xor(v1, 8, 16));
                v2 = fminf(v2, __shfl_xor(v2, 1, 16));   // cols 0..3 only
                v2 = fminf(v2, __shfl_xor(v2, 2, 16));
                ptsRaw = fminf(v1, __shfl(v2, 0, 16));
            } else {
                // polyline: min(forward = shift 0, reversed = acc2 col 4)
                float cf  = __shfl(acc1[j], 0, 16);
                float rev = __shfl(acc2[j], 4, 16);
                ptsRaw = fminf(cf, rev);
            }
            if (col == j) {
                int   row = j + 8 * half;            // C row M = j + 8*(lane>=16)
                int   nl  = nSub + row;
                float pts = ptsRaw * ninv;
                float l0 = s_logit[nl * 4 + 0];
                float l1 = s_logit[nl * 4 + 1];
                float l2 = s_logit[nl * 4 + 2];
                float mx = fmaxf(l0, fmaxf(l1, l2));
                float e0 = __expf(l0 - mx), e1 = __expf(l1 - mx), e2 = __expf(l2 - mx);
                float pl = (label == 0) ? e0 : (label == 1) ? e1 : e2;
                float cls = -pl / (e0 + e1 + e2);
                float cost = 2.0f * cls + 5.0f * pts;
                cost = (cost != cost) ? 1.0e6f
                                      : fminf(fmaxf(cost, -1.0e6f), 1.0e6f); // nan_to_num
                out[(size_t)(b * N_ + nBase + nl) * G_ + (gBase + gi)] = cost;
            }
        }
    }
}

extern "C" void kernel_launch(void* const* d_in, const int* in_sizes, int n_in,
                              void* d_out, int out_size, void* d_ws, size_t ws_size,
                              hipStream_t stream) {
    const float* pred_logits = (const float*)d_in[0];
    const float* pred_lines  = (const float*)d_in[1];
    const int*   gt_labels   = (const int*)d_in[2];
    const float* gt_lines    = (const float*)d_in[3];
    const int*   gt_masks    = (const int*)d_in[4];
    float* out = (float*)d_out;
    (void)in_sizes; (void)n_in; (void)out_size; (void)d_ws; (void)ws_size;

    dim3 grid(N_ / NTILE, G_ / GTILE, B_);   // 8 x 8 x 4 = 256 blocks ~ 1/WGP
    matcher_cost_kernel<<<grid, 256, 0, stream>>>(pred_logits, pred_lines,
                                                  gt_labels, gt_lines, gt_masks, out);
}
